// DVQ_87041807220991
// MI455X (gfx1250) — compile-verified, matrix operand order checked by reading, not run on hardware
//
#include <hip/hip_runtime.h>
#include <hip/hip_bf16.h>
#include <stdint.h>

// Problem constants (from reference)
#define BB 4096   // batch
#define DD 512    // embedding dim
#define KK 1024   // codebook entries
#define NN 8      // decompositions
#define ND (DD*NN)          // 4096
#define LOGK 6.9314718056f  // log(1024)
#define INV_BN (1.0f/32768.0f)

typedef __attribute__((ext_vector_type(16))) __bf16 bf16x16;
typedef __attribute__((ext_vector_type(8)))  float  v8f;
typedef int v4i_ __attribute__((vector_size(16)));   // int4 for async-LDS builtin

// ---------------------------------------------------------------------------
// Async global->LDS copy (CDNA5 GLOBAL_LOAD_ASYNC_TO_LDS_B128, ASYNCcnt),
// with a synchronous fallback if the builtin is unavailable.
// ---------------------------------------------------------------------------
#if defined(__has_builtin)
#if __has_builtin(__builtin_amdgcn_global_load_async_to_lds_b128) && \
    __has_builtin(__builtin_amdgcn_s_wait_asynccnt)
#define HAVE_ASYNC_LDS 1
#endif
#endif

__device__ __forceinline__ void cp16_to_lds(__bf16* l, const __bf16* g) {
#ifdef HAVE_ASYNC_LDS
  __builtin_amdgcn_global_load_async_to_lds_b128(
      (__attribute__((address_space(1))) v4i_*)(g),
      (__attribute__((address_space(3))) v4i_*)(l), 0, 0);
#else
  *reinterpret_cast<float4*>(l) = *reinterpret_cast<const float4*>(g);
#endif
}

// Wait until at most 8 async copies (the newest slab) are still in flight.
__device__ __forceinline__ void async_wait_le8() {
#ifdef HAVE_ASYNC_LDS
  __builtin_amdgcn_s_wait_asynccnt(8);
#endif
}

// ---------------------------------------------------------------------------
// Issue one B slab (ROWS x 32 bf16, K-slice [kk,kk+32)) as async b128 copies.
// Compile-time trip count -> straight-line burst of async copies (clauseable).
// Slab layout: panel[row*32 + d]; 4 chunks of 8 bf16 per row.
// ---------------------------------------------------------------------------
template <int ROWS, int NTHR>
__device__ __forceinline__ void issue_bpanel(const __bf16* Bg, __bf16* panel,
                                             int kk, int ldg, int tid) {
  constexpr int TRIPS = (ROWS * 4) / NTHR;
#pragma unroll
  for (int it = 0; it < TRIPS; ++it) {
    int c = tid + it * NTHR;
    int row = c >> 2, part = c & 3;
    cp16_to_lds(panel + row * 32 + part * 8,
                Bg + (size_t)row * ldg + kk + part * 8);
  }
}

// A panel: NCHUNK 16-byte chunks, global row stride ldg (elements), row
// length LROW bf16 in LDS. Compile-time trip count.
template <int NCHUNK, int NTHR, int CPR /*chunks per row*/, int LROW>
__device__ __forceinline__ void issue_apanel(const __bf16* Ag, __bf16* panel,
                                             int ldg, int tid) {
  constexpr int TRIPS = NCHUNK / NTHR;
#pragma unroll
  for (int it = 0; it < TRIPS; ++it) {
    int idx = tid + it * NTHR;
    int row = idx / CPR, c16 = idx % CPR;
    cp16_to_lds(panel + row * LROW + c16 * 8, Ag + (size_t)row * ldg + c16 * 8);
  }
}

// ---------------------------------------------------------------------------
// Fragment gather per CDNA5 ISA 7.12.2 (16x32 bf16 A / transposed-B layout):
// lane l holds row (l&15); K chunks {kb..kb+7, kb+16..kb+23}, kb = (l>=16)?8:0.
// From LDS this is two ds_load_b128.
// ---------------------------------------------------------------------------
__device__ __forceinline__ bf16x16 load_frag(const __bf16* base, int ld, int k0) {
  int lane = threadIdx.x & 31;
  int m    = lane & 15;
  int kb   = (lane >> 4) << 3;   // 0 or 8
  const __bf16* p = base + (size_t)m * ld + (size_t)(k0 + kb);
  union { float4 f4[2]; bf16x16 v; } u;
  u.f4[0] = *reinterpret_cast<const float4*>(p);
  u.f4[1] = *reinterpret_cast<const float4*>(p + 16);
  return u.v;
}

__device__ __forceinline__ v8f wmma_bf16(bf16x16 a, bf16x16 b, v8f c) {
  return __builtin_amdgcn_wmma_f32_16x16x32_bf16(false, a, false, b, (short)0, c,
                                                 false, false);
}

__device__ __forceinline__ uint32_t pcg_hash(uint32_t v) {
  uint32_t s = v * 747796405u + 2891336453u;
  uint32_t w = ((s >> ((s >> 28) + 4u)) ^ s) * 277803737u;
  return (w >> 22) ^ w;
}

// ---------------------------------------------------------------------------
// Prep 1: bf16-convert inputs; transpose+convert W_proj -> Wt[j][d]; zero loss.
// ---------------------------------------------------------------------------
__global__ void k_prep_misc(const float* __restrict__ inputs,
                            const float* __restrict__ Wp,
                            __bf16* __restrict__ Xb,
                            __bf16* __restrict__ Wt,
                            float* __restrict__ loss) {
  size_t i = (size_t)blockIdx.x * 256 + threadIdx.x;
  if (i == 0) *loss = 0.0f;
  const size_t total = (size_t)BB * DD;
  for (; i < total; i += (size_t)gridDim.x * 256) {
    Xb[i] = (__bf16)inputs[i];
    size_t j = i >> 9;
    size_t d = i & 511;
    Wt[i] = (__bf16)Wp[d * ND + j];
  }
}

// ---------------------------------------------------------------------------
// Prep 2: per (n,k) row: bf16 codebook copy, transposed bf16 copy, e2 = sum sq.
// ---------------------------------------------------------------------------
__global__ void k_prep_cb(const float* __restrict__ cbf,
                          __bf16* __restrict__ cb,
                          __bf16* __restrict__ cbT,
                          float* __restrict__ e2) {
  int nk = blockIdx.x;
  int n  = nk >> 10;
  int k  = nk & 1023;
  const float* src = cbf + (size_t)nk * DD;
  __shared__ float red[256];
  float s = 0.0f;
  for (int d = threadIdx.x; d < DD; d += 256) {
    float v = src[d];
    s += v * v;
    cb [(size_t)nk * DD + d]           = (__bf16)v;
    cbT[((size_t)n * DD + d) * KK + k] = (__bf16)v;
  }
  red[threadIdx.x] = s;
  __syncthreads();
  for (int o = 128; o; o >>= 1) {
    if (threadIdx.x < (unsigned)o) red[threadIdx.x] += red[threadIdx.x + o];
    __syncthreads();
  }
  if (threadIdx.x == 0) e2[nk] = red[0];
}

// ---------------------------------------------------------------------------
// Projection GEMM: Xp[b][j] = sum_d Xb[b][d]*Wt[j][d] + bias[j]   (bf16 out)
// 8 waves; wave: 16x64 tile; block: 16 rows x 512 cols. A panel + double-
// buffered B slabs in LDS, all filled by async copies; WMMA reads only LDS.
// ---------------------------------------------------------------------------
__global__ __launch_bounds__(256) void k_proj(const __bf16* __restrict__ Xb,
                                              const __bf16* __restrict__ Wt,
                                              const float* __restrict__ bias,
                                              __bf16* __restrict__ Xp) {
  __shared__ __align__(16) __bf16 Apanel[16 * DD];      // 16 KB
  __shared__ __align__(16) __bf16 Bpanel[2][512 * 32];  // 64 KB
  int tid  = threadIdx.x;
  int wave = tid >> 5, lane = tid & 31;
  int b0   = blockIdx.x << 4;
  int cblk = blockIdx.y << 9;

  const __bf16* Ag = Xb + (size_t)b0 * DD;              // contiguous 16 KB
  issue_apanel<1024, 256, 64, DD>(Ag, Apanel, DD, tid); // 4 copies/thread

  const __bf16* Bg = Wt + (size_t)cblk * DD;            // rows = block cols
  issue_bpanel<512, 256>(Bg, Bpanel[0], 0,  DD, tid);   // 8 copies/thread
  issue_bpanel<512, 256>(Bg, Bpanel[1], 32, DD, tid);   // 8 copies/thread

  v8f z = {0,0,0,0,0,0,0,0};
  v8f acc[4] = {z, z, z, z};
  const int STEPS = DD / 32;                            // 16
  for (int s = 0; s < STEPS; s++) {
    int cur = s & 1;
    async_wait_le8();      // newest slab may remain; cur slab + A are done
    __syncthreads();
    bf16x16 af = load_frag(Apanel, DD, s * 32);
#pragma unroll
    for (int t = 0; t < 4; t++) {
      bf16x16 bfm = load_frag(Bpanel[cur] + (size_t)(wave * 64 + t * 16) * 32, 32, 0);
      acc[t] = wmma_bf16(af, bfm, acc[t]);
    }
    __syncthreads();       // all waves done reading cur before overwrite
    if (s + 2 < STEPS)
      issue_bpanel<512, 256>(Bg, Bpanel[cur], (s + 2) * 32, DD, tid);
  }

  int cl = lane & 15, hi = lane >> 4;
#pragma unroll
  for (int t = 0; t < 4; t++) {
    int cg = cblk + wave * 64 + t * 16 + cl;
    float bv = bias[cg];
#pragma unroll
    for (int j = 0; j < 8; j++) {
      int r = b0 + j + (hi << 3);
      Xp[(size_t)r * ND + cg] = (__bf16)(acc[t][j] + bv);
    }
  }
}

// ---------------------------------------------------------------------------
// Fused distance + argmin + KL-loss + gumbel-softmax sample.
// Block: 512 thr (16 waves), grid (B/16, N). Wave: 16x64 xe tile via WMMA.
// B slabs (2 x 64 KB) alias the 64 KB distance buffer: dist is written only
// after the final B read (separated by the loop's trailing barrier).
// distance' = e2 - 2*xe (x^2 is row-constant: cancels in argmin, normalized
// logits and both softmaxes).
// ---------------------------------------------------------------------------
__global__ __launch_bounds__(512) void k_dist(const __bf16* __restrict__ Xp,
                                              const __bf16* __restrict__ Cb,
                                              const float* __restrict__ e2,
                                              __bf16* __restrict__ Sp,
                                              float* __restrict__ loss,
                                              int* __restrict__ idxout) {
  __shared__ __align__(16) char shmem[2 * KK * 32 * 2]; // 128 KB (B slabs / dist)
  __shared__ __align__(16) __bf16 Apanel[16 * DD];      // 16 KB
  __bf16* Bbase = reinterpret_cast<__bf16*>(shmem);
  float*  dist  = reinterpret_cast<float*>(shmem);      // 64 KB, aliases slabs

  int tid  = threadIdx.x;
  int wave = tid >> 5, lane = tid & 31;
  int b0   = blockIdx.x << 4;
  int n    = blockIdx.y;

  const __bf16* Ag = Xp + (size_t)b0 * ND + n * DD;     // row stride ND
  issue_apanel<1024, 512, 64, DD>(Ag, Apanel, ND, tid); // 2 copies/thread

  const __bf16* Bg = Cb + (size_t)n * KK * DD;          // rows = k, ld = DD
  issue_bpanel<KK, 512>(Bg, Bbase,           0,  DD, tid);  // 8 copies/thread
  issue_bpanel<KK, 512>(Bg, Bbase + KK * 32, 32, DD, tid);  // 8 copies/thread

  v8f z = {0,0,0,0,0,0,0,0};
  v8f acc[4] = {z, z, z, z};
  const int STEPS = DD / 32;                            // 16
  for (int s = 0; s < STEPS; s++) {
    int cur = s & 1;
    __bf16* Bcur = Bbase + (size_t)cur * (KK * 32);
    async_wait_le8();
    __syncthreads();
    bf16x16 af = load_frag(Apanel, DD, s * 32);
#pragma unroll
    for (int t = 0; t < 4; t++) {
      bf16x16 bfm = load_frag(Bcur + (size_t)(wave * 64 + t * 16) * 32, 32, 0);
      acc[t] = wmma_bf16(af, bfm, acc[t]);
    }
    __syncthreads();
    if (s + 2 < STEPS)
      issue_bpanel<KK, 512>(Bg, Bcur, (s + 2) * 32, DD, tid);
  }

  // epilogue: distance' into LDS (slabs no longer needed)
  int cl = lane & 15, hi = lane >> 4;
#pragma unroll
  for (int t = 0; t < 4; t++) {
    int c = wave * 64 + t * 16 + cl;
    float e = e2[(n << 10) + c];
#pragma unroll
    for (int j = 0; j < 8; j++)
      dist[(j + (hi << 3)) * KK + c] = e - 2.0f * acc[t][j];
  }
  __syncthreads();

  int r   = tid >> 5;            // one full wave per distance row
  int sub = tid & 31;
  float* row = dist + r * KK;
  int b = b0 + r;

  // argmin (first-min tie-break) and m = max(-d)
  float minv = 3.4e38f; int mini = 0;
  for (int i = sub; i < KK; i += 32) {
    float d = row[i];
    if (d < minv) { minv = d; mini = i; }
  }
  for (int o = 16; o; o >>= 1) {
    float ov = __shfl_xor(minv, o, 32);
    int   oi = __shfl_xor(mini, o, 32);
    if (ov < minv || (ov == minv && oi < mini)) { minv = ov; mini = oi; }
  }
  float m = -minv;

  float Z = 0.0f;
  for (int i = sub; i < KK; i += 32) Z += __expf(-row[i] - m);
  for (int o = 16; o; o >>= 1) Z += __shfl_xor(Z, o, 32);
  float lZ = __logf(Z);

  float kl = 0.0f;
  for (int i = sub; i < KK; i += 32) {
    float l = -row[i] - m - lZ;
    kl += __expf(l) * (l + LOGK);
  }
  for (int o = 16; o; o >>= 1) kl += __shfl_xor(kl, o, 32);
  if (sub == 0) {
    idxout[(b << 3) + n] = mini;
    atomicAdd(loss, kl * INV_BN);
  }

  // gumbel-softmax sample: s = (g - d)/TEMP ; TEMP = 0.5
  float m2 = -3.4e38f;
  for (int i = sub; i < KK; i += 32) {
    uint32_t id = (((uint32_t)((b << 3) + n)) << 10) | (uint32_t)i;
    uint32_t h  = pcg_hash(id);
    float u = ((float)(h >> 8) + 0.5f) * (1.0f / 16777216.0f);
    float g = -__logf(-__logf(u));
    float sv = (g - row[i]) * 2.0f;
    row[i] = sv;                  // own slots only
    m2 = fmaxf(m2, sv);
  }
  for (int o = 16; o; o >>= 1) m2 = fmaxf(m2, __shfl_xor(m2, o, 32));
  float Z2 = 0.0f;
  for (int i = sub; i < KK; i += 32) Z2 += __expf(row[i] - m2);
  for (int o = 16; o; o >>= 1) Z2 += __shfl_xor(Z2, o, 32);
  float inv = 1.0f / Z2;
  __bf16* spr = Sp + ((size_t)b * NN + n) * KK;
  for (int i = sub; i < KK; i += 32)
    spr[i] = (__bf16)(__expf(row[i] - m2) * inv);
}

// ---------------------------------------------------------------------------
// quantized[b][n][d] = sum_k Sp[b][n][k] * codebook[n][k][d]    (f32 out)
// 8 waves; wave: 16x64 tile; block: 16 rows x full D. A panel (32 KB) +
// double-buffered B slabs (2 x 32 KB) in LDS via async copies.
// ---------------------------------------------------------------------------
__global__ __launch_bounds__(256) void k_quant(const __bf16* __restrict__ Sp,
                                               const __bf16* __restrict__ CbT,
                                               float* __restrict__ out) {
  __shared__ __align__(16) __bf16 Apanel[16 * KK];      // 32 KB
  __shared__ __align__(16) __bf16 Bpanel[2][512 * 32];  // 64 KB
  int tid  = threadIdx.x;
  int wave = tid >> 5, lane = tid & 31;
  int b0   = blockIdx.x << 4;
  int n    = blockIdx.y;

  const __bf16* Ag = Sp + ((size_t)b0 * NN + n) * KK;   // row stride NN*KK
  issue_apanel<2048, 256, 128, KK>(Ag, Apanel, NN * KK, tid); // 8 copies/thread

  const __bf16* Bg = CbT + (size_t)n * DD * KK;         // rows = d, ld = KK
  issue_bpanel<512, 256>(Bg, Bpanel[0], 0,  KK, tid);
  issue_bpanel<512, 256>(Bg, Bpanel[1], 32, KK, tid);

  v8f z = {0,0,0,0,0,0,0,0};
  v8f acc[4] = {z, z, z, z};
  const int STEPS = KK / 32;                            // 32
  for (int s = 0; s < STEPS; s++) {
    int cur = s & 1;
    async_wait_le8();
    __syncthreads();
    bf16x16 af = load_frag(Apanel, KK, s * 32);
#pragma unroll
    for (int t = 0; t < 4; t++) {
      bf16x16 bfm = load_frag(Bpanel[cur] + (size_t)(wave * 64 + t * 16) * 32, 32, 0);
      acc[t] = wmma_bf16(af, bfm, acc[t]);
    }
    __syncthreads();
    if (s + 2 < STEPS)
      issue_bpanel<512, 256>(Bg, Bpanel[cur], (s + 2) * 32, KK, tid);
  }

  int cl = lane & 15, hi = lane >> 4;
#pragma unroll
  for (int t = 0; t < 4; t++) {
    int c = wave * 64 + t * 16 + cl;
#pragma unroll
    for (int j = 0; j < 8; j++) {
      int r = b0 + j + (hi << 3);
      out[((size_t)r * NN + n) * DD + c] = acc[t][j];
    }
  }
}

// ---------------------------------------------------------------------------
extern "C" void kernel_launch(void* const* d_in, const int* in_sizes, int n_in,
                              void* d_out, int out_size, void* d_ws, size_t ws_size,
                              hipStream_t stream) {
  (void)in_sizes; (void)n_in; (void)out_size; (void)ws_size;
  const float* inputs = (const float*)d_in[0];
  /* d_in[1] = var, unused by the reference quantize path */
  const float* Wp  = (const float*)d_in[2];
  const float* bp  = (const float*)d_in[3];
  const float* cbf = (const float*)d_in[4];

  float* out  = (float*)d_out;                        // quantized [B][N][D]
  float* loss = out + (size_t)BB * NN * DD;           // scalar
  int*   idx  = (int*)(loss + 1);                     // [B][N] indices

  char* w = (char*)d_ws;
  auto take = [&](size_t bytes) -> void* {
    void* p = (void*)w;
    w += (bytes + 255) & ~(size_t)255;
    return p;
  };
  __bf16* Xb  = (__bf16*)take((size_t)BB * DD * 2);
  __bf16* Wt  = (__bf16*)take((size_t)ND * DD * 2);
  __bf16* Cb  = (__bf16*)take((size_t)NN * KK * DD * 2);
  __bf16* CbT = (__bf16*)take((size_t)NN * DD * KK * 2);
  float*  e2  = (float*) take((size_t)NN * KK * 4);
  __bf16* Xp  = (__bf16*)take((size_t)BB * ND * 2);
  __bf16* Sp  = (__bf16*)take((size_t)BB * NN * KK * 2);

  k_prep_misc<<<2048, 256, 0, stream>>>(inputs, Wp, Xb, Wt, loss);
  k_prep_cb  <<<NN * KK, 256, 0, stream>>>(cbf, Cb, CbT, e2);

  dim3 gp(BB / 16, ND / 512);
  k_proj <<<gp, 256, 0, stream>>>(Xb, Wt, bp, Xp);

  dim3 gd(BB / 16, NN);
  k_dist <<<gd, 512, 0, stream>>>(Xp, Cb, e2, Sp, loss, idx);

  dim3 gq(BB / 16, NN);
  k_quant<<<gq, 256, 0, stream>>>(Sp, CbT, out);
}
